// MemoryBank_25821343384040
// MI455X (gfx1250) — matrix operand order repaired and checked by reading, not compiled, not run on hardware
//
#include <hip/hip_runtime.h>
#include <hip/hip_bf16.h>
#include <math.h>

// ---------------------------------------------------------------------------
// Problem constants (from reference)
// ---------------------------------------------------------------------------
#define NTOK   65536
#define DIM    256
#define LMEM   4
#define NHEAD  8
#define HDIM   32
#define HID    1024
#define TM     16            // tokens per workgroup
#define WAVES  8             // 256 threads / wave32
#define KCHUNK 8             // 8 x 32 = 256 K-elements software-pipelined

// bf16 weight workspace layout (element offsets)
#define O_WQ     0
#define O_WK     65536
#define O_WV     131072
#define O_WOUT   196608            // == 768*256, W_in is [Wq;Wk;Wv] contiguous
#define O_WSAVE  262144
#define O_WFC1   327680
#define O_WFC2   589824
#define W_TOTAL  851968

// output buffer layout (float elements), tuple order:
// new_emb (N,D), new_bank (N,L,D), new_mask (N,L), new_sp (N,)
#define OUT_EMB   0
#define OUT_BANK  (NTOK * DIM)
#define OUT_MASK  (OUT_BANK + NTOK * LMEM * DIM)
#define OUT_SP    (OUT_MASK + NTOK * LMEM)

typedef __attribute__((ext_vector_type(16))) __bf16 v16bf;
typedef __attribute__((ext_vector_type(8)))  __bf16 v8bf;
typedef __attribute__((ext_vector_type(8)))  float  v8f;

union FragBF { v16bf v; v8bf h[2]; };

// ---------------------------------------------------------------------------
// Weight conversion pre-pass: fp32 -> bf16 into workspace (L2 resident after)
// ---------------------------------------------------------------------------
__global__ __launch_bounds__(256) void convert_weights_kernel(
    const float* __restrict__ W_in,   // (768,256)
    const float* __restrict__ W_out,  // (256,256)
    const float* __restrict__ W_save, // (256,256)
    const float* __restrict__ W_fc1,  // (1024,256)
    const float* __restrict__ W_fc2,  // (256,1024)
    __bf16* __restrict__ ws)
{
  int idx = blockIdx.x * blockDim.x + threadIdx.x;
  if (idx >= W_TOTAL) return;
  float v;
  if      (idx < O_WOUT)  v = W_in  [idx];
  else if (idx < O_WSAVE) v = W_out [idx - O_WOUT];
  else if (idx < O_WFC1)  v = W_save[idx - O_WSAVE];
  else if (idx < O_WFC2)  v = W_fc1 [idx - O_WFC1];
  else                    v = W_fc2 [idx - O_WFC2];
  ws[idx] = (__bf16)v;
}

// ---------------------------------------------------------------------------
// Tiled bf16 GEMM core: D(16m x 16n tiles) = A(LDS, MxK) * W^T + bias
//   A: LDS bf16 row-major, stride lda (ISA 16-bit A layout: lane (l&15)=row,
//      K runs [kb,kb+8) and [kb+16,kb+24), kb=(l>>4)*8 -> two b128 LDS loads)
//   W: global bf16 (Nout x K) row-major; B[k][n]=W[n][k]; documented B layout:
//      lane (l&15)=col n, K runs [kb,kb+8) and [kb+16,kb+24), kb=(l>>4)*8
// K is software-pipelined in 256-wide chunks: all 16 global b128 loads and
// all 16 LDS b128 loads issue before the 8-WMMA chain, so the backend emits
// progressive s_wait_loadcnt/dscnt instead of a full stall per fragment.
// ---------------------------------------------------------------------------
__device__ __forceinline__ void gemm_bf16(
    const __bf16* __restrict__ A, int lda,
    const __bf16* __restrict__ W, int K,
    const float*  __restrict__ bias,
    int Mtiles, int Ntiles,
    float* __restrict__ outF, __bf16* __restrict__ outB, int ldo,
    bool relu, int wave, int lane)
{
  const int tiles = Mtiles * Ntiles;
  const int lm  = lane & 15;
  const int kbA = (lane >> 4) * 8;
  const int kbB = (lane >> 4) * 8;
  for (int t = wave; t < tiles; t += WAVES) {
    const int mt = t / Ntiles;
    const int nt = t - mt * Ntiles;
    v8f c = {0.f, 0.f, 0.f, 0.f, 0.f, 0.f, 0.f, 0.f};
    const __bf16* arow = A + (size_t)(mt * 16 + lm) * lda;
    const __bf16* brow = W + (size_t)(nt * 16 + lm) * K;
    for (int k0 = 0; k0 < K; k0 += 32 * KCHUNK) {
      FragBF a[KCHUNK], b[KCHUNK];
#pragma unroll
      for (int kk = 0; kk < KCHUNK; ++kk) {      // issue all global B loads
        const int k = k0 + kk * 32;
        b[kk].h[0] = *(const v8bf*)(brow + k + kbB);
        b[kk].h[1] = *(const v8bf*)(brow + k + kbB + 16);
      }
#pragma unroll
      for (int kk = 0; kk < KCHUNK; ++kk) {      // issue all LDS A loads
        const int k = k0 + kk * 32;
        a[kk].h[0] = *(const v8bf*)(arow + k + kbA);
        a[kk].h[1] = *(const v8bf*)(arow + k + kbA + 16);
      }
#pragma unroll
      for (int kk = 0; kk < KCHUNK; ++kk)        // WMMA chain
        c = __builtin_amdgcn_wmma_f32_16x16x32_bf16(
                false, a[kk].v, false, b[kk].v, (short)0, c, false, false);
    }
    const float bv   = bias[nt * 16 + lm];
    const int mbase  = mt * 16 + ((lane >> 4) << 3);
    const int n      = nt * 16 + lm;
#pragma unroll
    for (int r = 0; r < 8; ++r) {
      float val = c[r] + bv;
      if (relu) val = fmaxf(val, 0.f);
      const int m = mbase + r;
      if (outF) outF[m * ldo + n] = val;
      if (outB) outB[m * ldo + n] = (__bf16)val;
    }
  }
}

// ---------------------------------------------------------------------------
// Fused memory-bank transformer step, 16 tokens per workgroup
// ---------------------------------------------------------------------------
__global__ __launch_bounds__(256) void fused_membank_kernel(
    const float* __restrict__ emb,            // (N,D)
    const float* __restrict__ scores,         // (N,)
    const unsigned char* __restrict__ kpm,    // (N,L) bool
    const int*   __restrict__ save_period,    // (N,)
    const float* __restrict__ bank,           // (N,L,D)
    const float* __restrict__ b_save,
    const float* __restrict__ b_in,           // (3D,)
    const float* __restrict__ b_out,
    const float* __restrict__ ln1_g, const float* __restrict__ ln1_b,
    const float* __restrict__ b_fc1, const float* __restrict__ b_fc2,
    const float* __restrict__ ln2_g, const float* __restrict__ ln2_b,
    const __bf16* __restrict__ wbf,           // converted weights
    float* __restrict__ out)
{
  extern __shared__ char smem[];
  char* p = smem;
  float* xs    = (float*)p; p += TM * DIM * 4;            // original embedding
  float* embs  = (float*)p; p += TM * DIM * 4;            // post-LN1 embed
  float* tmp   = (float*)p; p += TM * DIM * 4;            // gemm f32 staging
  float* qs    = (float*)p; p += TM * DIM * 4;            // Q f32
  float* red   = (float*)p; p += TM * 16 * 4;
  float* red2  = (float*)p; p += TM * 16 * 4;
  float* meanv = (float*)p; p += TM * 4;
  float* rstdv = (float*)p; p += TM * 4;
  int*   kpmI  = (int*)p;   p += TM * LMEM * 4;
  int*   validI= (int*)p;   p += TM * 4;
  int*   savedI= (int*)p;   p += TM * 4;
  p = (char*)(((size_t)p + 255) & ~(size_t)255);
  __bf16* abf  = (__bf16*)p; p += TM * HID * 2;           // A-operand staging
  __bf16* hbf  = (__bf16*)p; p += TM * HID * 2;           // FFN hidden (relu)
  __bf16* mbbf = (__bf16*)p; p += TM * LMEM * DIM * 2;    // mem_bank bf16
  __bf16* ksb  = (__bf16*)p; p += TM * LMEM * DIM * 2;    // K bf16
  __bf16* vsb  = (__bf16*)p; p += TM * LMEM * DIM * 2;    // V bf16

  const int tid  = threadIdx.x;
  const int wave = tid >> 5;
  const int lane = tid & 31;
  const int t0   = blockIdx.x * TM;

  // ---- S0: stage inputs (fp32 residual copy + bf16 matmul copies) ----------
  for (int i = tid; i < TM * DIM; i += 256) {
    const float v = emb[(size_t)t0 * DIM + i];
    xs[i] = v;
    abf[(i >> 8) * HID + (i & 255)] = (__bf16)v;
  }
  for (int i = tid; i < TM * LMEM * DIM; i += 256)
    mbbf[i] = (__bf16)bank[(size_t)t0 * LMEM * DIM + i];
  if (tid < TM * LMEM) kpmI[tid] = (int)kpm[t0 * LMEM + tid];
  if (tid < TM)        validI[tid] = kpm[(t0 + tid) * LMEM + (LMEM - 1)] ? 0 : 1;
  __syncthreads();

  // ---- S1-S3: Q / K / V projections (WMMA) ---------------------------------
  gemm_bf16(abf,  HID, wbf + O_WQ, DIM, b_in,           1,       16, qs,  nullptr, DIM, false, wave, lane);
  gemm_bf16(mbbf, DIM, wbf + O_WK, DIM, b_in + DIM,     LMEM,    16, nullptr, ksb, DIM, false, wave, lane);
  gemm_bf16(mbbf, DIM, wbf + O_WV, DIM, b_in + 2 * DIM, LMEM,    16, nullptr, vsb, DIM, false, wave, lane);
  __syncthreads();

  // ---- S4: attention over L=4 slots (VALU; tiny) ---------------------------
  if (tid < TM * NHEAD) {
    const int t = tid >> 3, h = tid & 7;
    const float scale = 0.17677669529663687f;   // 1/sqrt(32)
    float lg[LMEM];
    const float* qp = qs + t * DIM + h * HDIM;
#pragma unroll
    for (int l = 0; l < LMEM; ++l) {
      const __bf16* kp = ksb + (t * LMEM + l) * DIM + h * HDIM;
      float acc = 0.f;
      for (int d = 0; d < HDIM; ++d) acc += qp[d] * (float)kp[d];
      acc *= scale;
      const bool masked = kpmI[t * LMEM + l] && validI[t];
      lg[l] = masked ? -__builtin_inff() : acc;
    }
    float mx = fmaxf(fmaxf(lg[0], lg[1]), fmaxf(lg[2], lg[3]));
    float e[LMEM], se = 0.f;
#pragma unroll
    for (int l = 0; l < LMEM; ++l) { e[l] = __expf(lg[l] - mx); se += e[l]; }
    const float inv = 1.f / se;
    for (int d = 0; d < HDIM; ++d) {
      float acc = 0.f;
#pragma unroll
      for (int l = 0; l < LMEM; ++l)
        acc += e[l] * (float)vsb[(t * LMEM + l) * DIM + h * HDIM + d];
      abf[t * HID + h * HDIM + d] = (__bf16)(acc * inv);
    }
  }
  __syncthreads();

  // ---- S5: attention output projection -------------------------------------
  gemm_bf16(abf, HID, wbf + O_WOUT, DIM, b_out, 1, 16, tmp, nullptr, DIM, false, wave, lane);
  __syncthreads();

  // ---- S6: LN1(x + attn_out) -> embs (f32) + abf (bf16) --------------------
  {
    const int t = tid >> 4, j = tid & 15;
    float s = 0.f, s2 = 0.f;
    for (int c = j; c < DIM; c += 16) {
      const float v = xs[t * DIM + c] + tmp[t * DIM + c];
      s += v; s2 += v * v;
    }
    red[t * 16 + j] = s; red2[t * 16 + j] = s2;
  }
  __syncthreads();
  if (tid < TM) {
    float s = 0.f, s2 = 0.f;
    for (int j = 0; j < 16; ++j) { s += red[tid * 16 + j]; s2 += red2[tid * 16 + j]; }
    const float m = s * (1.f / DIM);
    meanv[tid] = m;
    rstdv[tid] = rsqrtf(s2 * (1.f / DIM) - m * m + 1e-5f);
  }
  __syncthreads();
  for (int i = tid; i < TM * DIM; i += 256) {
    const int t = i >> 8, c = i & 255;
    const float v = xs[i] + tmp[i];
    const float y = (v - meanv[t]) * rstdv[t] * ln1_g[c] + ln1_b[c];
    embs[i] = y;
    abf[t * HID + c] = (__bf16)y;
  }
  __syncthreads();

  // ---- S7: FC1 (256 -> 1024) + ReLU ---------------------------------------
  gemm_bf16(abf, HID, wbf + O_WFC1, DIM, b_fc1, 1, 64, nullptr, hbf, HID, true, wave, lane);
  __syncthreads();

  // ---- S8: FC2 (1024 -> 256) ----------------------------------------------
  gemm_bf16(hbf, HID, wbf + O_WFC2, HID, b_fc2, 1, 16, tmp, nullptr, DIM, false, wave, lane);
  __syncthreads();

  // ---- S9: LN2(embed + ffn), valid-select, emit new_emb --------------------
  {
    const int t = tid >> 4, j = tid & 15;
    float s = 0.f, s2 = 0.f;
    for (int c = j; c < DIM; c += 16) {
      const float v = embs[t * DIM + c] + tmp[t * DIM + c];
      s += v; s2 += v * v;
    }
    red[t * 16 + j] = s; red2[t * 16 + j] = s2;
  }
  __syncthreads();
  if (tid < TM) {
    float s = 0.f, s2 = 0.f;
    for (int j = 0; j < 16; ++j) { s += red[tid * 16 + j]; s2 += red2[tid * 16 + j]; }
    const float m = s * (1.f / DIM);
    meanv[tid] = m;
    rstdv[tid] = rsqrtf(s2 * (1.f / DIM) - m * m + 1e-5f);
  }
  __syncthreads();
  for (int i = tid; i < TM * DIM; i += 256) {
    const int t = i >> 8, c = i & 255;
    const float v = embs[i] + tmp[i];
    float y = (v - meanv[t]) * rstdv[t] * ln2_g[c] + ln2_b[c];
    if (!validI[t]) y = xs[i];                 // invalid rows keep old embed
    out[OUT_EMB + (size_t)(t0 + t) * DIM + c] = y;
    abf[t * HID + c] = (__bf16)y;
  }
  __syncthreads();

  // ---- S10: save projection proj = new_emb @ W_save^T + b_save -------------
  gemm_bf16(abf, HID, wbf + O_WSAVE, DIM, b_save, 1, 16, tmp, nullptr, DIM, false, wave, lane);
  __syncthreads();

  // ---- S11: saved flags, new_sp, new_mask, new_bank ------------------------
  if (tid < TM) {
    const int t  = tid;
    const int sp = save_period[t0 + t];
    const int sv = (sp == 0) && (scores[t0 + t] > 0.f);
    savedI[t] = sv;
    int nsp = (sp > 0) ? sp - 1 : sp;
    if (sv) nsp = 3;                           // SAVE_PERIOD
    out[OUT_SP + t0 + t] = (float)nsp;
#pragma unroll
    for (int l = 0; l < LMEM; ++l) {
      int m;
      if (sv) m = (l < LMEM - 1) ? kpmI[t * LMEM + l + 1] : 0;
      else    m = kpmI[t * LMEM + l];
      out[OUT_MASK + (size_t)(t0 + t) * LMEM + l] = (float)m;
    }
  }
  __syncthreads();
  for (int i = tid; i < TM * LMEM * DIM; i += 256) {
    const int t = i >> 10;                     // i / (L*D)
    const int r = i & 1023;
    const int l = r >> 8, c = r & 255;
    const size_t gb = (size_t)(t0 + t) * LMEM * DIM;
    float v;
    if (savedI[t]) v = (l < LMEM - 1) ? bank[gb + (l + 1) * DIM + c] : tmp[t * DIM + c];
    else           v = bank[gb + l * DIM + c];
    out[OUT_BANK + gb + (size_t)l * DIM + c] = v;
  }
}

// ---------------------------------------------------------------------------
extern "C" void kernel_launch(void* const* d_in, const int* in_sizes, int n_in,
                              void* d_out, int out_size, void* d_ws, size_t ws_size,
                              hipStream_t stream) {
  const float* emb      = (const float*)d_in[0];
  const float* scores   = (const float*)d_in[1];
  const unsigned char* kpm = (const unsigned char*)d_in[2];
  const int*   sp       = (const int*)d_in[3];
  const float* bank     = (const float*)d_in[4];
  const float* W_save   = (const float*)d_in[5];
  const float* b_save   = (const float*)d_in[6];
  const float* W_in     = (const float*)d_in[7];
  const float* b_in     = (const float*)d_in[8];
  const float* W_out    = (const float*)d_in[9];
  const float* b_out    = (const float*)d_in[10];
  const float* ln1_g    = (const float*)d_in[11];
  const float* ln1_b    = (const float*)d_in[12];
  const float* W_fc1    = (const float*)d_in[13];
  const float* b_fc1    = (const float*)d_in[14];
  const float* W_fc2    = (const float*)d_in[15];
  const float* b_fc2    = (const float*)d_in[16];
  const float* ln2_g    = (const float*)d_in[17];
  const float* ln2_b    = (const float*)d_in[18];
  float*  out = (float*)d_out;
  __bf16* wbf = (__bf16*)d_ws;

  convert_weights_kernel<<<(W_TOTAL + 255) / 256, 256, 0, stream>>>(
      W_in, W_out, W_save, W_fc1, W_fc2, wbf);

  // dynamic LDS: 4 f32[16x256] + reductions/flags + 5 bf16 32KB buffers
  const size_t smem =
      (size_t)TM * DIM * 4 * 4 +                    // xs, embs, tmp, qs
      (size_t)TM * 16 * 4 * 2 + TM * 4 * 2 +        // red, red2, mean, rstd
      (size_t)TM * LMEM * 4 + TM * 4 * 2 + 256 +    // kpmI, validI, savedI, pad
      (size_t)TM * HID * 2 * 2 +                    // abf, hbf
      (size_t)TM * LMEM * DIM * 2 * 3;              // mbbf, ksb, vsb

  fused_membank_kernel<<<NTOK / TM, 256, smem, stream>>>(
      emb, scores, kpm, sp, bank,
      b_save, b_in, b_out, ln1_g, ln1_b, b_fc1, b_fc2, ln2_g, ln2_b,
      wbf, out);
}
